// AdjacencyMatrix_73452530696566
// MI455X (gfx1250) — matrix operand order repaired and checked by reading, not compiled, not run on hardware
//
#include <hip/hip_runtime.h>
#include <hip/hip_bf16.h>

// ---------------------------------------------------------------------------
// AdjacencyMatrix: G = softmax(theta(z) @ phi(z)^T) per (n,t)
// N=4, T=32, V=1024, C_IN=256, C_OUT=64
// Memory-bound: >=0.67 GB HBM traffic (z read + G write) vs ~26 GFLOP
// -> stream z once, write G once, bf16 WMMA (f32 accumulate), 32 MB bf16
//    intermediates in d_ws (fully L2-resident on the 192 MB L2).
// ---------------------------------------------------------------------------

#define NT     128          // N*T batches
#define V_DIM  1024
#define C_IN   256
#define C_OUT  64
#define ROWS   (NT * V_DIM) // 131072 projected rows
#define W_ELEMS (C_OUT * C_IN)

typedef __attribute__((ext_vector_type(16))) __bf16 v16bf;
typedef __attribute__((ext_vector_type(8)))  __bf16 v8bf;
typedef __attribute__((ext_vector_type(4)))  __bf16 v4bf;
typedef __attribute__((ext_vector_type(8)))  float  v8f;
typedef __attribute__((ext_vector_type(4)))  float  v4f;

// ---------------------------------------------------------------------------
// Kernel 0: one-shot f32 -> bf16 conversion of both weight matrices.
// 128 KB total; removes per-wave redundant v_cvt_pk_bf16_f32 in kernel 1.
// ---------------------------------------------------------------------------
__global__ void __launch_bounds__(256)
cvt_weights(const float* __restrict__ tw, const float* __restrict__ pw,
            __bf16* __restrict__ twb, __bf16* __restrict__ pwb)
{
    const int i = (blockIdx.x * 256 + threadIdx.x) * 4;
    if (i < W_ELEMS) {
        v4f t = *(const v4f*)(tw + i);
        v4f p = *(const v4f*)(pw + i);
        v4bf tb4, pb4;
#pragma unroll
        for (int k = 0; k < 4; ++k) {
            tb4[k] = (__bf16)t[k];
            pb4[k] = (__bf16)p[k];
        }
        *(v4bf*)(twb + i) = tb4;
        *(v4bf*)(pwb + i) = pb4;
    }
}

// ---------------------------------------------------------------------------
// Kernel 1: theta = z @ theta_w^T + theta_b ; phi = z @ phi_w^T + phi_b
// One wave -> 16 rows x 64 cols of BOTH projections (shares the z loads).
// A-frag (16x32 bf16): lane m(0..15) holds row m; lanes 0-15 take K 0..15
//   of the K-step, lanes 16-31 take K 16..31 (contiguous, same order as B).
// B-frag (32x16 bf16): lane n holds output column o=base+n; lanes 0-15 hold
//   K 0..15, lanes 16-31 hold K 16..31 -> contiguous 16 bf16 weights per lane.
// All 8 B-fragments of a K-step are preloaded into distinct registers so the
// 16 b128 loads fly as one clause and the 8 WMMAs drain with staggered waits.
// Epilogue: per-wave LDS transpose so global stores are full-wave b128
//   (512 B contiguous per instruction = 4 complete 128 B rows).
// ---------------------------------------------------------------------------
__global__ void __launch_bounds__(256)
proj_kernel(const float* __restrict__ z,
            const __bf16* __restrict__ twb, const float* __restrict__ tb,
            const __bf16* __restrict__ pwb, const float* __restrict__ pb,
            __bf16* __restrict__ theta, __bf16* __restrict__ phi)
{
    const int lane = threadIdx.x & 31;
    const int wave = threadIdx.x >> 5;
    const int ln   = lane & 15;
    const int half = lane >> 4;

    const long rowBlk = (long)blockIdx.x * 8 + wave;   // 8 waves/block
    const long r0     = rowBlk * 16;                   // 16 rows per wave

    v8f accT[4] = {{0.f,0.f,0.f,0.f,0.f,0.f,0.f,0.f}};
    v8f accP[4] = {{0.f,0.f,0.f,0.f,0.f,0.f,0.f,0.f}};
    accT[1] = accT[0]; accT[2] = accT[0]; accT[3] = accT[0];
    accP[1] = accP[0]; accP[2] = accP[0]; accP[3] = accP[0];

    const float* zrow = z + (r0 + ln) * C_IN;
    const __bf16* twl = twb + ln * C_IN + half * 16;   // per-lane weight base
    const __bf16* pwl = pwb + ln * C_IN + half * 16;

    for (int ks = 0; ks < C_IN / 32; ++ks) {
        const int k0 = ks * 32;

        // ---- A fragment: z rows, f32 -> bf16 (the only convert left) ----
        const float* zp = zrow + k0 + half * 16;
        v4f z0 = *(const v4f*)(zp);
        v4f z1 = *(const v4f*)(zp + 4);
        v4f z2 = *(const v4f*)(zp + 8);
        v4f z3 = *(const v4f*)(zp + 12);
        v16bf a;
#pragma unroll
        for (int i = 0; i < 4; ++i) {
            a[i]      = (__bf16)z0[i];
            a[4 + i]  = (__bf16)z1[i];
            a[8 + i]  = (__bf16)z2[i];
            a[12 + i] = (__bf16)z3[i];
        }

        // ---- preload ALL B fragments of this K-step (16x b128, one clause),
        //      then drain the 8 WMMAs with staggered loadcnt waits ----
        v16bf bt[4], bp[4];
#pragma unroll
        for (int ot = 0; ot < 4; ++ot) {
            bt[ot] = *(const v16bf*)(twl + ot * 16 * C_IN + k0);
            bp[ot] = *(const v16bf*)(pwl + ot * 16 * C_IN + k0);
        }
#pragma unroll
        for (int ot = 0; ot < 4; ++ot) {
            accT[ot] = __builtin_amdgcn_wmma_f32_16x16x32_bf16(
                false, a, false, bt[ot], (short)0, accT[ot], false, false);
            accP[ot] = __builtin_amdgcn_wmma_f32_16x16x32_bf16(
                false, a, false, bp[ot], (short)0, accP[ot], false, false);
        }
    }

    // ---- epilogue: bias + bf16, LDS transpose, coalesced b128 stores ----
    // C layout: VGPR j -> row j (+8 upper half), col = ot*16 + ln.
    __shared__ __bf16 tile[8][16][72];   // 72 = 64 + pad (bank spread), 18 KB

    const int rquad = lane >> 3;   // 0..3 : row within quad-row store group
    const int cseg  = lane & 7;    // 0..7 : 16 B segment within a 128 B row

    // ---- theta ----
    {
#pragma unroll
        for (int ot = 0; ot < 4; ++ot) {
            const int o  = ot * 16 + ln;
            const float bT = tb[o];
#pragma unroll
            for (int j = 0; j < 8; ++j)
                tile[wave][j + 8 * half][o] = (__bf16)(accT[ot][j] + bT);
        }
        // per-wave only: DS ops are in-order within a wave, no barrier needed
#pragma unroll
        for (int it = 0; it < 4; ++it) {
            const int rloc = it * 4 + rquad;
            v8bf q = *(const v8bf*)(&tile[wave][rloc][cseg * 8]);
            *(v8bf*)(theta + (r0 + rloc) * C_OUT + cseg * 8) = q;
        }
    }
    // ---- phi (reuse the same LDS tile) ----
    {
#pragma unroll
        for (int ot = 0; ot < 4; ++ot) {
            const int o  = ot * 16 + ln;
            const float bP = pb[o];
#pragma unroll
            for (int j = 0; j < 8; ++j)
                tile[wave][j + 8 * half][o] = (__bf16)(accP[ot][j] + bP);
        }
#pragma unroll
        for (int it = 0; it < 4; ++it) {
            const int rloc = it * 4 + rquad;
            v8bf q = *(const v8bf*)(&tile[wave][rloc][cseg * 8]);
            *(v8bf*)(phi + (r0 + rloc) * C_OUT + cseg * 8) = q;
        }
    }
}

// ---------------------------------------------------------------------------
// Kernel 2: S = theta @ phi^T per batch, row softmax, write G (f32).
// Workgroup = 4 waves, handles 16 rows x 1024 cols of one batch.
// Wave w owns columns [w*256, w*256+256): 16 accumulator tiles in registers.
// Softmax: shfl_xor reduce within 16-lane halves, 512 B LDS across 4 waves.
// ---------------------------------------------------------------------------
__global__ void __launch_bounds__(128)
affinity_softmax_kernel(const __bf16* __restrict__ theta,
                        const __bf16* __restrict__ phi,
                        float* __restrict__ out)
{
    const int lane = threadIdx.x & 31;
    const int wave = threadIdx.x >> 5;   // 0..3
    const int ln   = lane & 15;
    const int half = lane >> 4;

    const long b  = blockIdx.y;          // batch (n*t)
    const int  rt = blockIdx.x;          // 16-row tile within V

    const __bf16* thB = theta + (b * V_DIM + rt * 16) * C_OUT;
    const __bf16* phB = phi   + b * V_DIM * C_OUT;

    // ---- A fragments: theta rows, K = 0..63 in two 32-wide steps ----
    const __bf16* tr = thB + ln * C_OUT;
    v8bf t0 = *(const v8bf*)(tr +  0 + half * 16);
    v8bf t1 = *(const v8bf*)(tr +  8 + half * 16);
    v8bf t2 = *(const v8bf*)(tr + 32 + half * 16);
    v8bf t3 = *(const v8bf*)(tr + 40 + half * 16);
    v16bf a0, a1;
#pragma unroll
    for (int i = 0; i < 8; ++i) {
        a0[i] = t0[i]; a0[8 + i] = t1[i];
        a1[i] = t2[i]; a1[8 + i] = t3[i];
    }

    // ---- 16 column tiles of 16: S accumulation (all in registers) ----
    v8f acc[16];
#pragma unroll
    for (int ct = 0; ct < 16; ++ct) {
        const int w = wave * 256 + ct * 16 + ln;       // phi row = S column
        const __bf16* pr = phB + (long)w * C_OUT;
        v16bf b0 = *(const v16bf*)(pr +  0 + half * 16);
        v16bf b1 = *(const v16bf*)(pr + 32 + half * 16);
        v8f c = {0.f,0.f,0.f,0.f,0.f,0.f,0.f,0.f};
        c = __builtin_amdgcn_wmma_f32_16x16x32_bf16(
            false, a0, false, b0, (short)0, c, false, false);
        c = __builtin_amdgcn_wmma_f32_16x16x32_bf16(
            false, a1, false, b1, (short)0, c, false, false);
        acc[ct] = c;
    }

    // ---- softmax over the full 1024-wide row ----
    __shared__ float redMax[16][4];
    __shared__ float redSum[16][4];

    float rmax[8];
#pragma unroll
    for (int j = 0; j < 8; ++j) {
        float m = -3.402823466e38f;
#pragma unroll
        for (int ct = 0; ct < 16; ++ct) m = fmaxf(m, acc[ct][j]);
        for (int off = 1; off < 16; off <<= 1)     // stays inside each half
            m = fmaxf(m, __shfl_xor(m, off, 32));
        rmax[j] = m;
    }
    if (ln == 0) {
#pragma unroll
        for (int j = 0; j < 8; ++j) redMax[j + half * 8][wave] = rmax[j];
    }
    __syncthreads();

    float gmax[8];
#pragma unroll
    for (int j = 0; j < 8; ++j) {
        const int row = j + half * 8;
        float m = fmaxf(fmaxf(redMax[row][0], redMax[row][1]),
                        fmaxf(redMax[row][2], redMax[row][3]));
        gmax[j] = m;
    }

    float rsum[8];
#pragma unroll
    for (int j = 0; j < 8; ++j) {
        float s = 0.f;
#pragma unroll
        for (int ct = 0; ct < 16; ++ct) {
            float e = __expf(acc[ct][j] - gmax[j]);
            acc[ct][j] = e;
            s += e;
        }
        for (int off = 1; off < 16; off <<= 1)
            s += __shfl_xor(s, off, 32);
        rsum[j] = s;
    }
    if (ln == 0) {
#pragma unroll
        for (int j = 0; j < 8; ++j) redSum[j + half * 8][wave] = rsum[j];
    }
    __syncthreads();

    float rinv[8];
#pragma unroll
    for (int j = 0; j < 8; ++j) {
        const int row = j + half * 8;
        const float s = redSum[row][0] + redSum[row][1]
                      + redSum[row][2] + redSum[row][3];
        rinv[j] = __frcp_rn(s);
    }

    // ---- write G: out[b][row][col], f32 (64 B contiguous per half-wave,
    //      adjacent ct fill adjacent 64 B chunks -> full-line combining) ----
    float* outB = out + b * (long)V_DIM * V_DIM;
#pragma unroll
    for (int j = 0; j < 8; ++j) {
        const int row = rt * 16 + j + half * 8;
        float* orow = outB + (long)row * V_DIM + wave * 256 + ln;
#pragma unroll
        for (int ct = 0; ct < 16; ++ct)
            orow[ct * 16] = acc[ct][j] * rinv[j];
    }
}

// ---------------------------------------------------------------------------
extern "C" void kernel_launch(void* const* d_in, const int* in_sizes, int n_in,
                              void* d_out, int out_size, void* d_ws, size_t ws_size,
                              hipStream_t stream)
{
    const float* z  = (const float*)d_in[0];
    const float* tw = (const float*)d_in[1];
    const float* tb = (const float*)d_in[2];
    const float* pw = (const float*)d_in[3];
    const float* pb = (const float*)d_in[4];
    float* out = (float*)d_out;

    // workspace layout: theta(16MB) | phi(16MB) | twb(32KB) | pwb(32KB)
    __bf16* theta = (__bf16*)d_ws;
    __bf16* phi   = theta + (size_t)ROWS * C_OUT;
    __bf16* twb   = phi   + (size_t)ROWS * C_OUT;
    __bf16* pwb   = twb   + (size_t)W_ELEMS;

    // Kernel 0: convert 2 x 64x256 f32 weights to bf16 once
    cvt_weights<<<dim3((W_ELEMS / 4 + 255) / 256), dim3(256), 0, stream>>>(
        tw, pw, twb, pwb);

    // Kernel 1: 131072 rows / 16 rows-per-wave = 8192 waves; 8 waves/block
    proj_kernel<<<dim3(ROWS / (16 * 8)), dim3(256), 0, stream>>>(
        z, twb, tb, pwb, pb, theta, phi);

    // Kernel 2: grid = (64 row tiles, 128 batches), 4 waves/block
    affinity_softmax_kernel<<<dim3(V_DIM / 16, NT), dim3(128), 0, stream>>>(
        theta, phi, out);
}
